// USTGA_87522843557923
// MI455X (gfx1250) — matrix operand order repaired and checked by reading, not compiled, not run on hardware
//
#include <hip/hip_runtime.h>

typedef _Float16 f16;
typedef __attribute__((ext_vector_type(16))) _Float16 v16h;
typedef __attribute__((ext_vector_type(8)))  float    v8f;

#define B_   8
#define T_   4
#define N_   2000
#define D_   64
#define NT3  6000     /* S*N */
#define TT_  8        /* T_TOT */
#define HZ_  12
#define HID_ 128

static __device__ __forceinline__ v8f wmma16(v16h a, v16h b, v8f c) {
  // v_wmma_f32_16x16x32_f16  D = A*B + C  (f32 accumulate)
  return __builtin_amdgcn_wmma_f32_16x16x32_f16(false, a, false, b, (short)0, c, false, false);
}

union FragView { v16h v; float4 q[2]; };
union PairU { unsigned u; f16 h[2]; };

// A-fragment (16x32 f16, row-major source): lane row = lane&15,
// K(e) = e + (e>=8 ? 8 : 0) + ((lane&16) ? 8 : 0)  -> two contiguous 16B chunks.
static __device__ __forceinline__ v16h load_afrag(const f16* row_k0, int lane) {
  int off = (lane & 16) ? 8 : 0;
  FragView r;
  r.q[0] = *reinterpret_cast<const float4*>(row_k0 + off);
  r.q[1] = *reinterpret_cast<const float4*>(row_k0 + 16 + off);
  return r.v;
}

// B-fragment (32x16) from a K-major source (this lane's column stored as a
// contiguous K run): K(e) = e + (e>=8 ? 8 : 0) + ((lane&16) ? 8 : 0) -> two 16B chunks.
static __device__ __forceinline__ v16h load_bfrag_kcontig(const f16* row_k0, int lane) {
  int off = (lane & 16) ? 8 : 0;
  FragView r;
  r.q[0] = *reinterpret_cast<const float4*>(row_k0 + off);
  r.q[1] = *reinterpret_cast<const float4*>(row_k0 + 16 + off);
  return r.v;
}

// ---------------------------------------------------------------- kernel 1
// h = relu(x @ W_first + b);  z[:,0:4]=h (f16);  xe0 = h + emb0 (f16)
__global__ void k_first(const float* __restrict__ x, const float* __restrict__ W,
                        const float* __restrict__ bias, const float* __restrict__ emb0,
                        f16* __restrict__ z, f16* __restrict__ xe0) {
  int idx = blockIdx.x * blockDim.x + threadIdx.x;   // over B*T*N
  if (idx >= B_ * T_ * N_) return;
  int n = idx % N_;
  int bt = idx / N_;
  int b = bt / T_, t = bt % T_;
  float x0 = x[idx * 3 + 0], x1 = x[idx * 3 + 1], x2 = x[idx * 3 + 2];
  f16* zrow = z + ((size_t)(b * TT_ + t) * N_ + n) * D_;
  f16* xrow = xe0 + ((size_t)bt * N_ + n) * D_;
  const float* erow = emb0 + n * D_;
  for (int d = 0; d < D_; ++d) {
    float h = fmaxf(x0 * W[d] + x1 * W[64 + d] + x2 * W[128 + d] + bias[d], 0.f);
    zrow[d] = (f16)h;
    xrow[d] = (f16)(h + erow[d]);
  }
}

// ---------------------------------------------------------------- kernel 2
// a_f16[i][n<2000][m] = f16(adj * mask)  (only first N rows are ever used)
__global__ void k_prep_a(const float* __restrict__ adj, const float* __restrict__ masks,
                         f16* __restrict__ a) {
  size_t idx = (size_t)blockIdx.x * blockDim.x + threadIdx.x;
  size_t per_layer = (size_t)N_ * NT3;
  if (idx >= 2 * per_layer) return;
  int i = (int)(idx / per_layer);
  size_t r = idx % per_layer;
  int n = (int)(r / NT3), m = (int)(r % NT3);
  size_t src = (size_t)i * NT3 * NT3 + (size_t)n * NT3 + m;
  __builtin_prefetch(adj + src + 2048, 0, 1);    // global_prefetch_b8 on stream
  __builtin_prefetch(masks + src + 2048, 0, 1);
  a[idx] = (f16)(adj[src] * masks[src]);
}

// ---------------------------------------------------------------- kernel 3
// f16 + transpose prep of all B-side weights so WMMA B-fragments are K-contiguous:
//   w1t[hz][j][k]  <- fc1_w[hz][k][j]    (12*128*512)
//   awt[iw][j][k]  <- agg_W[iw][k][j]    (4*128*64)
//   wqt[j][k]      <- Wq[k][j]           (64*64)
//   wvt[j][k]      <- Wv[k][j]           (64*64)
__global__ void k_prep_w(const float* __restrict__ fc1w, const float* __restrict__ aggW,
                         const float* __restrict__ Wq, const float* __restrict__ Wv,
                         f16* __restrict__ w1t, f16* __restrict__ awt,
                         f16* __restrict__ wqt, f16* __restrict__ wvt) {
  int idx = blockIdx.x * blockDim.x + threadIdx.x;
  const int N1 = HZ_ * HID_ * 512;        // 786432
  const int N2 = 4 * HID_ * 64;           // 32768
  const int N3 = 64 * 64;                 // 4096
  if (idx < N1) {
    int hz = idx / (HID_ * 512);
    int r = idx % (HID_ * 512);
    int j = r / 512, k = r % 512;
    w1t[idx] = (f16)fc1w[((size_t)hz * 512 + k) * HID_ + j];
  } else if (idx < N1 + N2) {
    int o = idx - N1;
    int iw = o / (HID_ * 64);
    int r = o % (HID_ * 64);
    int j = r / 64, k = r % 64;
    awt[o] = (f16)aggW[((size_t)iw * 64 + k) * HID_ + j];
  } else if (idx < N1 + N2 + N3) {
    int o = idx - N1 - N2;
    int j = o / 64, k = o % 64;
    wqt[o] = (f16)Wq[k * 64 + j];
  } else if (idx < N1 + N2 + 2 * N3) {
    int o = idx - N1 - N2 - N3;
    int j = o / 64, k = o % 64;
    wvt[o] = (f16)Wv[k * 64 + j];
  }
}

// ---------------------------------------------------------------- kernel 4
// q = h@Wq+bq ; v = h@Wv+bv  — WMMA over 4000 row tiles of 16
__global__ void __launch_bounds__(256) k_qv(const f16* __restrict__ z,
                                            const f16* __restrict__ wqt, const float* __restrict__ bq,
                                            const f16* __restrict__ wvt, const float* __restrict__ bvv,
                                            f16* __restrict__ q, f16* __restrict__ v) {
  int lane = threadIdx.x & 31;
  int wv = threadIdx.x >> 5;
  int tile = blockIdx.x * 8 + wv;
  if (tile >= (B_ * T_ * N_) / 16) return;            // wave-uniform
  int col15 = lane & 15, half8 = (lane & 16) ? 8 : 0;
  int r = tile * 16 + col15;                          // A row for this lane
  int b = r / (T_ * N_);
  int rem = r % (T_ * N_);
  int t = rem / N_, n = rem % N_;
  const f16* hrow = z + ((size_t)(b * TT_ + t) * N_ + n) * D_;
  v16h a0 = load_afrag(hrow, lane);
  v16h a1 = load_afrag(hrow + 32, lane);
#pragma unroll
  for (int dt = 0; dt < 4; ++dt) {
    int col = dt * 16 + col15;
    const f16* wqr = wqt + (size_t)col * 64;
    const f16* wvr = wvt + (size_t)col * 64;
    v8f cq = {}, cv = {};
    cq = wmma16(a0, load_bfrag_kcontig(wqr, lane), cq);
    cq = wmma16(a1, load_bfrag_kcontig(wqr + 32, lane), cq);
    cv = wmma16(a0, load_bfrag_kcontig(wvr, lane), cv);
    cv = wmma16(a1, load_bfrag_kcontig(wvr + 32, lane), cv);
    float bqc = bq[col], bvc = bvv[col];
#pragma unroll
    for (int j = 0; j < 8; ++j) {
      size_t row = (size_t)tile * 16 + j + half8;
      q[row * D_ + col] = (f16)(cq[j] + bqc);
      v[row * D_ + col] = (f16)(cv[j] + bvc);
    }
  }
}

// ---------------------------------------------------------------- kernel 5
// flash attention per (b,t): scores = q v^T / 8, softmax, out = P v; xe1 = out + emb1
__global__ void __launch_bounds__(160) k_attn(const f16* __restrict__ q,
                                              const f16* __restrict__ v,
                                              const float* __restrict__ emb1,
                                              f16* __restrict__ xe1) {
  __shared__ __align__(16) f16 sV[32][64];     // row-major (QK^T: K=d contiguous)
  __shared__ __align__(16) f16 sVT[64][40];    // K-major for PV B-fragments
  __shared__ __align__(16) f16 sP[5][16][32];
  int lane = threadIdx.x & 31, wv = threadIdx.x >> 5;
  int bt = blockIdx.y;                       // 0..31
  int ntile = blockIdx.x * 5 + wv;           // 0..124 exact
  int col15 = lane & 15, half8 = (lane & 16) ? 8 : 0;
  const f16* qb = q + (size_t)bt * N_ * D_;
  const f16* vb = v + (size_t)bt * N_ * D_;
  const f16* qrow = qb + ((size_t)ntile * 16 + col15) * D_;
  v16h a0 = load_afrag(qrow, lane);
  v16h a1 = load_afrag(qrow + 32, lane);
  v8f o[4] = {};
  float mrow[8], lrow[8];
#pragma unroll
  for (int j = 0; j < 8; ++j) { mrow[j] = -3e38f; lrow[j] = 0.f; }

  for (int m0 = 0; m0 < N_; m0 += 32) {
    for (int idx = threadIdx.x; idx < 1024; idx += 160) {   // 32x64 as f16x2 pairs
      int rr = idx >> 5, cc = (idx & 31) * 2;
      int m = m0 + rr;
      PairU p; p.u = 0u;
      if (m < N_) p.u = *reinterpret_cast<const unsigned*>(&vb[(size_t)m * D_ + cc]);
      *reinterpret_cast<unsigned*>(&sV[rr][cc]) = p.u;
      sVT[cc][rr] = p.h[0];
      sVT[cc + 1][rr] = p.h[1];
    }
    __syncthreads();
    // S tiles (2 x 16x16), K=64
    v8f s[2];
#pragma unroll
    for (int st = 0; st < 2; ++st) {
      const f16* vrow = &sV[st * 16 + col15][0];   // v^T: K contiguous
      v8f c = {};
      c = wmma16(a0, load_bfrag_kcontig(vrow, lane), c);
      c = wmma16(a1, load_bfrag_kcontig(vrow + 32, lane), c);
      s[st] = c;
    }
#pragma unroll
    for (int j = 0; j < 8; ++j) {
      float x0 = s[0][j] * 0.125f, x1 = s[1][j] * 0.125f;
      if (m0 + col15 >= N_) x0 = -3e38f;
      if (m0 + 16 + col15 >= N_) x1 = -3e38f;
      float mm = fmaxf(x0, x1);
      for (int xm = 1; xm < 16; xm <<= 1) mm = fmaxf(mm, __shfl_xor(mm, xm, 16));
      float mnew = fmaxf(mrow[j], mm);
      float sc = __expf(mrow[j] - mnew);
      mrow[j] = mnew;
      float p0 = __expf(x0 - mnew), p1 = __expf(x1 - mnew);
      float rs = p0 + p1;
      for (int xm = 1; xm < 16; xm <<= 1) rs += __shfl_xor(rs, xm, 16);
      lrow[j] = lrow[j] * sc + rs;
      sP[wv][j + half8][col15] = (f16)p0;
      sP[wv][j + half8][16 + col15] = (f16)p1;
#pragma unroll
      for (int dt = 0; dt < 4; ++dt) o[dt][j] *= sc;
    }
    // PV: P (16x32) as A-fragment from wave-local LDS (DS ops in-order per wave)
    v16h pa = load_afrag(&sP[wv][col15][0], lane);
#pragma unroll
    for (int dt = 0; dt < 4; ++dt)
      o[dt] = wmma16(pa, load_bfrag_kcontig(&sVT[dt * 16 + col15][0], lane), o[dt]);
    __syncthreads();
  }
  f16* xout = xe1 + (size_t)bt * N_ * D_;
#pragma unroll
  for (int j = 0; j < 8; ++j) {
    int n = ntile * 16 + j + half8;
    float inv = 1.f / lrow[j];
#pragma unroll
    for (int dt = 0; dt < 4; ++dt) {
      int col = dt * 16 + col15;
      xout[(size_t)n * D_ + col] = (f16)(o[dt][j] * inv + emb1[n * D_ + col]);
    }
  }
}

// ---------------------------------------------------------------- kernel 6
// agg[n,d] = sum_m a[n,m] * xe[b, w+m/N, m%N, d] for n<2000; then fused
// h2 = agg @ W[i][w] + b (WMMA), GLU, write z slot 4+2i+w.
__global__ void __launch_bounds__(160) k_agg(const f16* __restrict__ a,
                                             const f16* __restrict__ xe,
                                             const f16* __restrict__ awt,
                                             const float* __restrict__ ab,
                                             f16* __restrict__ z) {
  __shared__ __align__(16) f16 sXeT[64][40];   // K-major xe tile
  __shared__ __align__(16) f16 sAgg[5][16][64];
  int lane = threadIdx.x & 31, wv = threadIdx.x >> 5;
  int b = blockIdx.y;
  int iw = blockIdx.z, i = iw >> 1, w = iw & 1;
  int ntile = blockIdx.x * 5 + wv;
  int col15 = lane & 15, half8 = (lane & 16) ? 8 : 0;
  const f16* arow = a + ((size_t)i * N_ + ntile * 16 + col15) * NT3;
  const f16* xeb = xe + (size_t)i * (B_ * T_ * N_ * D_) + (size_t)b * (T_ * N_ * D_);
  v8f c[4] = {};
  for (int m0 = 0; m0 < NT3; m0 += 32) {               // 188 steps, tail padded
    for (int idx = threadIdx.x; idx < 1024; idx += 160) {  // 32x64 as f16x2 pairs
      int rr = idx >> 5, cc = (idx & 31) * 2;
      int m = m0 + rr;
      PairU p; p.u = 0u;
      if (m < NT3) {
        int s = m / N_, nn = m % N_;
        p.u = *reinterpret_cast<const unsigned*>(&xeb[((size_t)(w + s) * N_ + nn) * D_ + cc]);
      }
      sXeT[cc][rr] = p.h[0];
      sXeT[cc + 1][rr] = p.h[1];
    }
    __syncthreads();
    v16h af = load_afrag(arow + m0, lane);             // padded past 6000 (slack)
#pragma unroll
    for (int dt = 0; dt < 4; ++dt)
      c[dt] = wmma16(af, load_bfrag_kcontig(&sXeT[dt * 16 + col15][0], lane), c[dt]);
    __syncthreads();
  }
  // stage agg tile as f16 for second GEMM
#pragma unroll
  for (int dt = 0; dt < 4; ++dt)
#pragma unroll
    for (int j = 0; j < 8; ++j)
      sAgg[wv][j + half8][dt * 16 + col15] = (f16)c[dt][j];
  __syncthreads();
  const f16* grow = &sAgg[wv][col15][0];
  v16h g0 = load_afrag(grow, lane);
  v16h g1 = load_afrag(grow + 32, lane);
  v8f t2[8];
#pragma unroll
  for (int jt = 0; jt < 8; ++jt) {
    const f16* awr = awt + ((size_t)iw * HID_ + jt * 16 + col15) * 64;  // K-major
    v8f cc = {};
    cc = wmma16(g0, load_bfrag_kcontig(awr, lane), cc);
    cc = wmma16(g1, load_bfrag_kcontig(awr + 32, lane), cc);
    t2[jt] = cc;
  }
  const float* abrow = ab + iw * HID_;
  f16* zout = z + (size_t)(b * TT_ + 4 + iw) * N_ * D_;
#pragma unroll
  for (int jt = 0; jt < 4; ++jt) {
    int col = jt * 16 + col15;
    float bval = abrow[col], bgat = abrow[col + 64];
#pragma unroll
    for (int j = 0; j < 8; ++j) {
      float valp = t2[jt][j] + bval;
      float gate = t2[jt + 4][j] + bgat;
      float g = valp / (1.f + __expf(-gate));          // GLU
      int n = ntile * 16 + j + half8;
      zout[(size_t)n * D_ + col] = (f16)g;
    }
  }
}

// ---------------------------------------------------------------- kernel 7
// fused fc1(relu)+fc2 per (b, horizon): out[b,hz,n]
__global__ void __launch_bounds__(160) k_fc(const f16* __restrict__ z,
                                            const f16* __restrict__ w1t,
                                            const float* __restrict__ b1,
                                            const float* __restrict__ w2,
                                            const float* __restrict__ b2,
                                            float* __restrict__ out) {
  __shared__ __align__(16) f16 sWT[HID_][40];  // K-major fc1 weight chunk
  int lane = threadIdx.x & 31, wv = threadIdx.x >> 5;
  int by = blockIdx.y;
  int b = by / HZ_, hz = by % HZ_;
  int ntile = blockIdx.x * 5 + wv;
  int col15 = lane & 15, half8 = (lane & 16) ? 8 : 0;
  const f16* w1h = w1t + (size_t)hz * HID_ * 512;   // [j][k] K-major
  v8f c[8] = {};
  for (int ks = 0; ks < 16; ++ks) {
    int k0 = ks * 32;
    for (int idx = threadIdx.x; idx < 2048; idx += 160) {  // 128x32 as f16x2 pairs
      int j = idx >> 4, kk = (idx & 15) * 2;
      *reinterpret_cast<unsigned*>(&sWT[j][kk]) =
          *reinterpret_cast<const unsigned*>(&w1h[(size_t)j * 512 + k0 + kk]);
    }
    __syncthreads();
    int tt = k0 >> 6, dbase = k0 & 63;                 // zf[b,n,k] = z[b,tt,n,d]
    const f16* zrow = z + ((size_t)(b * TT_ + tt) * N_ + ntile * 16 + col15) * D_ + dbase;
    v16h af = load_afrag(zrow, lane);
#pragma unroll
    for (int jt = 0; jt < 8; ++jt)
      c[jt] = wmma16(af, load_bfrag_kcontig(&sWT[jt * 16 + col15][0], lane), c[jt]);
    __syncthreads();
  }
  float acc[8] = {0.f, 0.f, 0.f, 0.f, 0.f, 0.f, 0.f, 0.f};
#pragma unroll
  for (int jt = 0; jt < 8; ++jt) {
    int col = jt * 16 + col15;
    float bb = b1[hz * HID_ + col], ww = w2[hz * HID_ + col];
#pragma unroll
    for (int j = 0; j < 8; ++j) acc[j] += fmaxf(c[jt][j] + bb, 0.f) * ww;
  }
#pragma unroll
  for (int j = 0; j < 8; ++j)
    for (int xm = 1; xm < 16; xm <<= 1) acc[j] += __shfl_xor(acc[j], xm, 16);
  if (col15 == 0) {
    float bias2 = b2[hz];
#pragma unroll
    for (int j = 0; j < 8; ++j) {
      int n = ntile * 16 + j + half8;
      out[((size_t)b * HZ_ + hz) * N_ + n] = acc[j] + bias2;
    }
  }
}

// ---------------------------------------------------------------- launch
extern "C" void kernel_launch(void* const* d_in, const int* in_sizes, int n_in,
                              void* d_out, int out_size, void* d_ws, size_t ws_size,
                              hipStream_t stream) {
  const float* x    = (const float*)d_in[0];
  const float* adj  = (const float*)d_in[1];
  const float* mk   = (const float*)d_in[2];
  const float* spa  = (const float*)d_in[3];
  const float* Wf   = (const float*)d_in[4];
  const float* bf   = (const float*)d_in[5];
  const float* Wq   = (const float*)d_in[6];
  const float* bq   = (const float*)d_in[7];
  const float* Wv   = (const float*)d_in[8];
  const float* bvv  = (const float*)d_in[9];
  const float* aW   = (const float*)d_in[10];
  const float* aB   = (const float*)d_in[11];
  const float* f1w  = (const float*)d_in[12];
  const float* f1b  = (const float*)d_in[13];
  const float* f2w  = (const float*)d_in[14];
  const float* f2b  = (const float*)d_in[15];
  float* out = (float*)d_out;

  char* ws = (char*)d_ws;
  size_t off = 0;
  auto alloc = [&](size_t bytes) -> void* {
    void* p = ws + off;
    off += (bytes + 255) & ~(size_t)255;
    return p;
  };
  f16* z   = (f16*)alloc(sizeof(f16) * (size_t)B_ * TT_ * N_ * D_);
  f16* xe  = (f16*)alloc(sizeof(f16) * (size_t)2 * B_ * T_ * N_ * D_);
  f16* qb  = (f16*)alloc(sizeof(f16) * (size_t)B_ * T_ * N_ * D_);
  f16* vb  = (f16*)alloc(sizeof(f16) * (size_t)B_ * T_ * N_ * D_);
  f16* af  = (f16*)alloc(sizeof(f16) * ((size_t)2 * N_ * NT3 + 64)); // +tail slack
  f16* w1t = (f16*)alloc(sizeof(f16) * (size_t)HZ_ * HID_ * 512);
  f16* awt = (f16*)alloc(sizeof(f16) * (size_t)4 * HID_ * 64);
  f16* wqt = (f16*)alloc(sizeof(f16) * (size_t)64 * 64);
  f16* wvt = (f16*)alloc(sizeof(f16) * (size_t)64 * 64);

  k_first<<<(B_ * T_ * N_ + 255) / 256, 256, 0, stream>>>(x, Wf, bf, spa, z, xe);
  size_t na = (size_t)2 * N_ * NT3;
  k_prep_a<<<(int)((na + 255) / 256), 256, 0, stream>>>(adj, mk, af);
  int nprep = HZ_ * HID_ * 512 + 4 * HID_ * 64 + 2 * 64 * 64;
  k_prep_w<<<(nprep + 255) / 256, 256, 0, stream>>>(f1w, aW, Wq, Wv, w1t, awt, wqt, wvt);
  k_qv<<<500, 256, 0, stream>>>(z, wqt, bq, wvt, bvv, qb, vb);
  k_attn<<<dim3(25, 32), 160, 0, stream>>>(qb, vb, spa + (size_t)N_ * D_,
                                           xe + (size_t)B_ * T_ * N_ * D_);
  k_agg<<<dim3(25, B_, 4), 160, 0, stream>>>(af, xe, awt, aB, z);
  k_fc<<<dim3(25, B_ * HZ_), 160, 0, stream>>>(z, w1t, f1b, f2w, f2b, out);
}
